// BahdanauAttention_73358041416406
// MI455X (gfx1250) — matrix operand order repaired and checked by reading, not compiled, not run on hardware
//
#include <hip/hip_runtime.h>
#include <hip/hip_bf16.h>

// Problem constants (match reference)
#define NB    128      // batch
#define NT    2048     // time steps
#define NH    512      // ENC_H == DEC_H
#define NA    128      // ATTN
#define TILE_T 64      // rows per tile
#define ASTR  528      // bf16 elems per sA row (512 + 16 pad -> conflict-free-ish)
#define WSTR  528      // bf16 elems per sW row

typedef __attribute__((ext_vector_type(16))) __bf16 v16bf;
typedef __attribute__((ext_vector_type(8)))  float  v8f;
typedef __attribute__((ext_vector_type(4)))  float  f4;
typedef __attribute__((ext_vector_type(4)))  unsigned int u4;
typedef __attribute__((ext_vector_type(4)))  unsigned short us4;

union Frag { v16bf v; u4 q[2]; };

__device__ __forceinline__ unsigned short f2bf(float f) {
  union { float f; unsigned int u; } x; x.f = f;
  unsigned int r = x.u + 0x7FFFu + ((x.u >> 16) & 1u);   // round-to-nearest-even
  return (unsigned short)(r >> 16);
}

__global__ __launch_bounds__(256, 1)
void bahdanau_fused_wmma(const float* __restrict__ enc,
                         const float* __restrict__ dec_h,
                         const float* __restrict__ W_enc,
                         const float* __restrict__ W_dec,
                         const float* __restrict__ v_w,
                         float* __restrict__ out_ctx,   // [NB, NH]
                         float* __restrict__ out_wts) { // [NB, NT]
  extern __shared__ char smem[];
  unsigned short* sW   = (unsigned short*)smem;          // [NA][WSTR]  bf16 W_enc (B operand, col-major over K)
  unsigned short* sA   = sW + NA * WSTR;                 // [TILE_T][ASTR] bf16 enc tile (A operand, row-major)
  float* s_all  = (float*)(sA + TILE_T * ASTR);          // [NT] raw scores
  float* s_tile = s_all + NT;                            // [TILE_T] tile scores -> tile exp-weights
  float* s_dp   = s_tile + TILE_T;                       // [NA] dec_proj
  float* s_v    = s_dp + NA;                             // [NA] v vector
  float* s_run  = s_v + NA;                              // [0]=run_max [1]=run_sum [2]=rescale

  const int tid  = threadIdx.x;
  const int lane = tid & 31;
  const int wave = tid >> 5;
  const int b    = blockIdx.x;
  const float* enc_b = enc + (size_t)b * NT * NH;

  // ---------- per-block setup ----------
  {
    // W_enc f32 -> bf16 into LDS; memory layout [a][h] already matches B operand needs
    const f4* w4 = (const f4*)W_enc;                     // 128*128 f4
    #pragma unroll
    for (int it = 0; it < 64; ++it) {
      int idx = tid + it * 256;
      int row = idx >> 7, c4 = idx & 127;
      f4 v = w4[idx];
      us4 o; o.x = f2bf(v.x); o.y = f2bf(v.y); o.z = f2bf(v.z); o.w = f2bf(v.w);
      *(us4*)(sW + row * WSTR + c4 * 4) = o;
    }
  }
  if (tid < NA) {
    // dec_proj[a] = W_dec[a,:] . dec_h[b,:]  (tiny, f32 VALU)
    const f4* wd = (const f4*)(W_dec + tid * NH);
    const f4* dh = (const f4*)(dec_h + (size_t)b * NH);
    float acc = 0.f;
    for (int i = 0; i < NH / 4; ++i) {
      f4 a = wd[i], d = dh[i];
      acc += a.x * d.x + a.y * d.y + a.z * d.z + a.w * d.w;
    }
    s_dp[tid] = acc;
    s_v[tid]  = v_w[tid];
  }
  if (tid == 0) { s_run[0] = -3.4e38f; s_run[1] = 0.f; s_run[2] = 0.f; }
  __syncthreads();

  const int m_blk  = wave & 3;          // which 16-row strip of the 64-row tile
  const int n_base = (wave >> 2) * 4;   // which four 16-col strips of the 128 A-dims
  const int half   = lane >> 4;
  const int l15    = lane & 15;

  float c0 = 0.f, c1 = 0.f;             // this thread's two context columns
  const int h0 = tid * 2;

  for (int tt = 0; tt < NT / TILE_T; ++tt) {
    const int t0 = tt * TILE_T;

    // ---- phase A: stream tile (f32, non-temporal) -> bf16 in LDS ----
    {
      const f4* e4 = (const f4*)(enc_b + (size_t)t0 * NH);  // 64*128 f4
      #pragma unroll
      for (int it = 0; it < 32; ++it) {
        int idx = tid + it * 256;
        int row = idx >> 7, c4 = idx & 127;
        f4 v = __builtin_nontemporal_load(&e4[idx]);
        us4 o; o.x = f2bf(v.x); o.y = f2bf(v.y); o.z = f2bf(v.z); o.w = f2bf(v.w);
        *(us4*)(sA + row * ASTR + c4 * 4) = o;
      }
      if (tid < TILE_T) s_tile[tid] = 0.f;
    }
    __syncthreads();

    // ---- phase B: WMMA enc_proj, fused tanh + v-dot -> tile scores ----
    v8f acc[4];
    #pragma unroll
    for (int nb = 0; nb < 4; ++nb)
      #pragma unroll
      for (int j = 0; j < 8; ++j) acc[nb][j] = 0.f;

    // A-fragment lane mapping (16-bit A 16x32): lanes 0-15 rows m, K {k0..k0+7, k0+16..k0+23};
    // lanes 16-31 same rows, K shifted by 8.
    const unsigned short* aRow = sA + (m_blk * 16 + l15) * ASTR + half * 8;
    #pragma unroll 4
    for (int kb = 0; kb < NH / 32; ++kb) {
      Frag af;
      af.q[0] = *(const u4*)(aRow + kb * 32);        // K: k0 + half*8 .. +7
      af.q[1] = *(const u4*)(aRow + kb * 32 + 16);   // K: k0 + 16 + half*8 .. +7
      #pragma unroll
      for (int nb = 0; nb < 4; ++nb) {
        // B-fragment (16-bit B 32x16): lane n holds column n; K {half*8..+7, 16+half*8..+7} (matches A)
        const unsigned short* bCol =
            sW + ((n_base + nb) * 16 + l15) * WSTR + kb * 32 + half * 8;
        Frag bf_;
        bf_.q[0] = *(const u4*)(bCol);
        bf_.q[1] = *(const u4*)(bCol + 16);
        acc[nb] = __builtin_amdgcn_wmma_f32_16x16x32_bf16(
            false, af.v, false, bf_.v, (short)0, acc[nb], false, false);
      }
    }

    // epilogue: score[t] = sum_a v[a] * tanh(enc_proj[t][a] + dec_proj[a])
    float partial[8];
    #pragma unroll
    for (int j = 0; j < 8; ++j) partial[j] = 0.f;
    #pragma unroll
    for (int nb = 0; nb < 4; ++nb) {
      int a_idx = (n_base + nb) * 16 + l15;
      float dp = s_dp[a_idx], vv = s_v[a_idx];
      #pragma unroll
      for (int j = 0; j < 8; ++j)
        partial[j] += vv * tanhf(acc[nb][j] + dp);
    }
    #pragma unroll
    for (int j = 0; j < 8; ++j) {
      float r = partial[j];
      r += __shfl_xor(r, 1); r += __shfl_xor(r, 2);
      r += __shfl_xor(r, 4); r += __shfl_xor(r, 8);
      if (l15 == 0)   // lane 0 -> row j, lane 16 -> row j+8 (C/D layout)
        atomicAdd(&s_tile[m_blk * 16 + j + half * 8], r);   // ds_add_f32
    }
    __syncthreads();

    // ---- phase C: online softmax bookkeeping (wave 0) ----
    if (wave == 0) {
      float sc0 = s_tile[lane], sc1 = s_tile[lane + 32];
      s_all[t0 + lane]      = sc0;
      s_all[t0 + lane + 32] = sc1;
      float mx = fmaxf(sc0, sc1);
      mx = fmaxf(mx, __shfl_xor(mx, 1));
      mx = fmaxf(mx, __shfl_xor(mx, 2));
      mx = fmaxf(mx, __shfl_xor(mx, 4));
      mx = fmaxf(mx, __shfl_xor(mx, 8));
      mx = fmaxf(mx, __shfl_xor(mx, 16));
      float old_m = s_run[0];
      float new_m = fmaxf(old_m, mx);
      float e0 = __expf(sc0 - new_m), e1 = __expf(sc1 - new_m);
      s_tile[lane]      = e0;
      s_tile[lane + 32] = e1;
      float ts = e0 + e1;
      ts += __shfl_xor(ts, 1); ts += __shfl_xor(ts, 2);
      ts += __shfl_xor(ts, 4); ts += __shfl_xor(ts, 8);
      ts += __shfl_xor(ts, 16);
      if (lane == 0) {
        float scale = __expf(old_m - new_m);
        s_run[0] = new_m;
        s_run[1] = s_run[1] * scale + ts;
        s_run[2] = scale;
      }
    }
    __syncthreads();

    // ---- phase D: rescale + accumulate context from bf16 tile in LDS ----
    float scale = s_run[2];
    c0 *= scale; c1 *= scale;
    const unsigned short* colp = sA + h0;
    #pragma unroll 8
    for (int t = 0; t < TILE_T; ++t) {
      float w = s_tile[t];
      unsigned int pk = *(const unsigned int*)(colp + t * ASTR);
      union { unsigned int u; float f; } lo, hi;
      lo.u = pk << 16;
      hi.u = pk & 0xFFFF0000u;
      c0 += w * lo.f;
      c1 += w * hi.f;
    }
    if (tt + 1 < NT / TILE_T)   // hint next tile toward L2 (global_prefetch_b8)
      __builtin_prefetch(enc_b + (size_t)(t0 + TILE_T) * NH + (size_t)tid * 128, 0, 0);
    __syncthreads();
  }

  // ---------- finalize ----------
  float inv = 1.0f / s_run[1];
  float fm  = s_run[0];
  out_ctx[(size_t)b * NH + h0]     = c0 * inv;
  out_ctx[(size_t)b * NH + h0 + 1] = c1 * inv;
  for (int t = tid; t < NT; t += 256)
    out_wts[(size_t)b * NT + t] = __expf(s_all[t] - fm) * inv;
}

extern "C" void kernel_launch(void* const* d_in, const int* in_sizes, int n_in,
                              void* d_out, int out_size, void* d_ws, size_t ws_size,
                              hipStream_t stream) {
  (void)in_sizes; (void)n_in; (void)out_size; (void)d_ws; (void)ws_size;
  const float* enc   = (const float*)d_in[0];   // [128, 2048, 512]
  const float* dec_h = (const float*)d_in[1];   // [128, 512]
  const float* W_enc = (const float*)d_in[2];   // [128, 512]
  const float* W_dec = (const float*)d_in[3];   // [128, 512]
  const float* v_w   = (const float*)d_in[4];   // [1, 128]
  float* out_ctx = (float*)d_out;               // context first: 128*512
  float* out_wts = out_ctx + (size_t)NB * NH;   // then weights: 128*2048

  constexpr size_t SMEM =
      (size_t)(NA * WSTR + TILE_T * ASTR) * sizeof(unsigned short) +
      (size_t)(NT + TILE_T + NA + NA + 4) * sizeof(float);   // ~207 KB < 320 KB

  bahdanau_fused_wmma<<<dim3(NB), dim3(256), SMEM, stream>>>(
      enc, dec_h, W_enc, W_dec, v_w, out_ctx, out_wts);
}
